// SwiGLUMLP_15805479649498
// MI455X (gfx1250) — matrix operand order repaired and checked by reading, not compiled, not run on hardware
//
#include <hip/hip_runtime.h>
#include <stddef.h>
#include <stdint.h>

// Problem constants (match reference)
#define BB 4
#define SS 2048
#define HH 2048
#define II 8192
#define GG 128
#define MM (BB * SS)   // 8192 tokens

typedef __attribute__((ext_vector_type(16))) __bf16 v16bf;
typedef __attribute__((ext_vector_type(8)))  float  v8f;
typedef __attribute__((ext_vector_type(8)))  unsigned int v8u;

// ---------------------------------------------------------------------------
// float -> bf16 (round-to-nearest-even)
// ---------------------------------------------------------------------------
__device__ __forceinline__ unsigned short f2bf(float f) {
    unsigned u = __builtin_bit_cast(unsigned, f);
    unsigned r = u + 0x7FFFu + ((u >> 16) & 1u);
    return (unsigned short)(r >> 16);
}

// ---------------------------------------------------------------------------
// Fragment layouts (1 KB per 16x32 bf16 fragment, 32 B per lane).
//
// A-fragment (ISA 16-bit A 16x32): lane = khalf*16 + (m%16), khalf=(k%16)/8.
//   lane bytes [0:16)  = A[m, 16*(k/16) + khalf*8 + 0..7]   (low 16-chunk)
//   lane bytes [16:32) = next 16-chunk, same 8-offset.
// Stored A-major: fragIdx = mtile*(K/32) + ktile.
//
// B-fragment (K x 16, row striped across lanes): lane = ((k%16)/8)*16 + (n%16),
//   lane bytes [0:16) = B[ktile*32 + khalf*8 + 0..7, n], [16:32) = +16 (same as A).
// Stored B-major: fragIdx = ntile*(K/32) + ktile.
// ---------------------------------------------------------------------------

// One wave per quantization group of 128 along the input dim.
// w: [out_f, in_f] fp32 row-major -> wq: B-fragment-major bf16 (sign * scale).
__global__ void __launch_bounds__(256)
quant_pack_w_kernel(const float* __restrict__ w, unsigned short* __restrict__ wq,
                    int out_f, int in_f) {
    int gwave = (int)((blockIdx.x * blockDim.x + threadIdx.x) >> 5);
    int lane  = (int)(threadIdx.x & 31);
    int gpr   = in_f >> 7;            // groups per output row
    int n     = gwave / gpr;          // output row
    int g     = gwave - n * gpr;      // group within row
    if (n >= out_f) return;

    const float4 wv = *(const float4*)(w + (size_t)n * in_f + (size_t)g * 128 + lane * 4);
    float s = fabsf(wv.x) + fabsf(wv.y) + fabsf(wv.z) + fabsf(wv.w);
    #pragma unroll
    for (int off = 16; off > 0; off >>= 1)
        s += __shfl_xor(s, off, 32);
    float scale = s * (1.0f / 128.0f);

    float q0 = (wv.x > 0.f) ? scale : ((wv.x < 0.f) ? -scale : 0.f);
    float q1 = (wv.y > 0.f) ? scale : ((wv.y < 0.f) ? -scale : 0.f);
    float q2 = (wv.z > 0.f) ? scale : ((wv.z < 0.f) ? -scale : 0.f);
    float q3 = (wv.w > 0.f) ? scale : ((wv.w < 0.f) ? -scale : 0.f);

    // Destination inside B-fragment layout
    int kbase  = g * 128 + lane * 4;
    int kt     = kbase >> 5;
    int k_in   = kbase & 31;
    int lane_d = (((k_in >> 3) & 1) << 4) + (n & 15);
    size_t base = ((size_t)((n >> 4) * (in_f >> 5) + kt)) * 1024
                + (size_t)lane_d * 32 + (size_t)((k_in >> 4) * 8 + (k_in & 7)) * 2;

    uint2 pk;
    pk.x = (unsigned)f2bf(q0) | ((unsigned)f2bf(q1) << 16);
    pk.y = (unsigned)f2bf(q2) | ((unsigned)f2bf(q3) << 16);
    *(uint2*)((char*)wq + base) = pk;
}

// x: [M, H] fp32 row-major -> xq: A-fragment-major bf16. One thread per 8-K chunk.
__global__ void __launch_bounds__(256)
pack_x_kernel(const float* __restrict__ x, unsigned short* __restrict__ xq) {
    const int K = HH;
    int tid = (int)(blockIdx.x * blockDim.x + threadIdx.x);
    int m = tid >> 8;            // K/8 = 256 chunks per row
    int c = tid & 255;
    int k = c << 3;

    const float4* p = (const float4*)(x + (size_t)m * K + k);
    float4 f0 = p[0];
    float4 f1 = p[1];

    int mt = m >> 4, m_in = m & 15;
    int kt = k >> 5, k_in = k & 31;
    int chunk16 = k_in >> 4;
    int khalf   = (k_in >> 3) & 1;
    int lane    = khalf * 16 + m_in;
    size_t base = ((size_t)(mt * (K >> 5) + kt)) * 1024
                + (size_t)lane * 32 + (size_t)chunk16 * 16;

    uint4 pk;
    pk.x = (unsigned)f2bf(f0.x) | ((unsigned)f2bf(f0.y) << 16);
    pk.y = (unsigned)f2bf(f0.z) | ((unsigned)f2bf(f0.w) << 16);
    pk.z = (unsigned)f2bf(f1.x) | ((unsigned)f2bf(f1.y) << 16);
    pk.w = (unsigned)f2bf(f1.z) | ((unsigned)f2bf(f1.w) << 16);
    *(uint4*)((char*)xq + base) = pk;
}

__device__ __forceinline__ v16bf load_frag(const char* p, int lane) {
    return __builtin_bit_cast(v16bf, *(const v8u*)(p + (size_t)lane * 32));
}

__device__ __forceinline__ v8f wmma_bf16(v16bf a, v16bf b, v8f c) {
    return __builtin_amdgcn_wmma_f32_16x16x32_bf16(false, a, false, b,
                                                   (short)0, c, false, false);
}

// ---------------------------------------------------------------------------
// GEMM1: gate/up fused.  Block = 128(M) x 64(N), 8 waves in 4x2.
// Wave tile = 32x32 for BOTH gate and up (8 accumulators).
// Epilogue: h = silu(gate)*up, stored row-major bf16 [M, I].
// ---------------------------------------------------------------------------
__global__ void __launch_bounds__(256)
gemm_gate_up_kernel(const unsigned short* __restrict__ xq,
                    const unsigned short* __restrict__ wgq,
                    const unsigned short* __restrict__ wuq,
                    unsigned short* __restrict__ hbuf) {
    constexpr int K  = HH;
    constexpr int KT = K / 32;      // 64 k-steps
    constexpr int N  = II;

    int lane = (int)(threadIdx.x & 31);
    int wv   = (int)(threadIdx.x >> 5);
    int wm   = wv >> 1;             // 0..3
    int wn   = wv & 1;              // 0..1
    int m0   = (int)blockIdx.x * 128 + wm * 32;
    int n0   = (int)blockIdx.y * 64  + wn * 32;

    const size_t rowStride = (size_t)KT * 1024;   // bytes per 16-row fragment strip
    const char* aBase = (const char*)xq  + (size_t)(m0 >> 4) * rowStride;
    const char* gBase = (const char*)wgq + (size_t)(n0 >> 4) * rowStride;
    const char* uBase = (const char*)wuq + (size_t)(n0 >> 4) * rowStride;

    v8f z = {0.f, 0.f, 0.f, 0.f, 0.f, 0.f, 0.f, 0.f};
    v8f accg[2][2], accu[2][2];
    #pragma unroll
    for (int i = 0; i < 2; ++i)
        #pragma unroll
        for (int j = 0; j < 2; ++j) { accg[i][j] = z; accu[i][j] = z; }

    for (int kt = 0; kt < KT; ++kt) {
        size_t ko = (size_t)kt * 1024;
        v16bf a0 = load_frag(aBase + ko, lane);
        v16bf a1 = load_frag(aBase + rowStride + ko, lane);
        v16bf g0 = load_frag(gBase + ko, lane);
        v16bf g1 = load_frag(gBase + rowStride + ko, lane);
        v16bf u0 = load_frag(uBase + ko, lane);
        v16bf u1 = load_frag(uBase + rowStride + ko, lane);

        accg[0][0] = wmma_bf16(a0, g0, accg[0][0]);
        accg[0][1] = wmma_bf16(a0, g1, accg[0][1]);
        accg[1][0] = wmma_bf16(a1, g0, accg[1][0]);
        accg[1][1] = wmma_bf16(a1, g1, accg[1][1]);
        accu[0][0] = wmma_bf16(a0, u0, accu[0][0]);
        accu[0][1] = wmma_bf16(a0, u1, accu[0][1]);
        accu[1][0] = wmma_bf16(a1, u0, accu[1][0]);
        accu[1][1] = wmma_bf16(a1, u1, accu[1][1]);
    }

    // SiLU(gate) * up epilogue.  C layout: VGPR r -> m = r + (lane/16)*8, n = lane%16.
    #pragma unroll
    for (int fm = 0; fm < 2; ++fm) {
        #pragma unroll
        for (int fn = 0; fn < 2; ++fn) {
            v8f g = accg[fm][fn];
            v8f u = accu[fm][fn];
            int mb = m0 + fm * 16 + (lane >> 4) * 8;
            int nn = n0 + fn * 16 + (lane & 15);
            #pragma unroll
            for (int r = 0; r < 8; ++r) {
                float gv = g[r];
                float hv = (gv / (1.0f + __expf(-gv))) * u[r];
                hbuf[(size_t)(mb + r) * N + nn] = f2bf(hv);
            }
        }
    }
}

// ---------------------------------------------------------------------------
// GEMM2: out = h . wd^T.  Block = 128(M) x 64(N), wave tile 32x32.
// A fragments gathered on the fly from row-major bf16 h.
// ---------------------------------------------------------------------------
__global__ void __launch_bounds__(256)
gemm_down_kernel(const unsigned short* __restrict__ hbuf,
                 const unsigned short* __restrict__ wdq,
                 float* __restrict__ out) {
    constexpr int K  = II;
    constexpr int KT = K / 32;      // 256 k-steps
    constexpr int N  = HH;

    int lane = (int)(threadIdx.x & 31);
    int wv   = (int)(threadIdx.x >> 5);
    int wm   = wv >> 1;
    int wn   = wv & 1;
    int m0   = (int)blockIdx.x * 128 + wm * 32;
    int n0   = (int)blockIdx.y * 64  + wn * 32;

    // A gather: lane holds m = m0 + fm*16 + (lane&15), k-chunks at khalf*8 and 16+khalf*8.
    const unsigned short* aRow0 =
        hbuf + (size_t)(m0 + (lane & 15)) * K + (lane >> 4) * 8;
    const unsigned short* aRow1 = aRow0 + (size_t)16 * K;

    const size_t rowStride = (size_t)KT * 1024;
    const char* dBase = (const char*)wdq + (size_t)(n0 >> 4) * rowStride;

    v8f z = {0.f, 0.f, 0.f, 0.f, 0.f, 0.f, 0.f, 0.f};
    v8f acc[2][2];
    #pragma unroll
    for (int i = 0; i < 2; ++i)
        #pragma unroll
        for (int j = 0; j < 2; ++j) acc[i][j] = z;

    for (int kt = 0; kt < KT; ++kt) {
        int kb = kt * 32;
        union { uint4 q[2]; v16bf v; } A0, A1;
        A0.q[0] = *(const uint4*)(aRow0 + kb);
        A0.q[1] = *(const uint4*)(aRow0 + kb + 16);
        A1.q[0] = *(const uint4*)(aRow1 + kb);
        A1.q[1] = *(const uint4*)(aRow1 + kb + 16);

        size_t ko = (size_t)kt * 1024;
        v16bf b0 = load_frag(dBase + ko, lane);
        v16bf b1 = load_frag(dBase + rowStride + ko, lane);

        acc[0][0] = wmma_bf16(A0.v, b0, acc[0][0]);
        acc[0][1] = wmma_bf16(A0.v, b1, acc[0][1]);
        acc[1][0] = wmma_bf16(A1.v, b0, acc[1][0]);
        acc[1][1] = wmma_bf16(A1.v, b1, acc[1][1]);
    }

    #pragma unroll
    for (int fm = 0; fm < 2; ++fm) {
        #pragma unroll
        for (int fn = 0; fn < 2; ++fn) {
            v8f c = acc[fm][fn];
            int mb = m0 + fm * 16 + (lane >> 4) * 8;
            int nn = n0 + fn * 16 + (lane & 15);
            #pragma unroll
            for (int r = 0; r < 8; ++r)
                out[(size_t)(mb + r) * N + nn] = c[r];
        }
    }
}

// ---------------------------------------------------------------------------
// Launch
// ---------------------------------------------------------------------------
extern "C" void kernel_launch(void* const* d_in, const int* in_sizes, int n_in,
                              void* d_out, int out_size, void* d_ws, size_t ws_size,
                              hipStream_t stream) {
    const float* x      = (const float*)d_in[0];   // [B,S,H]
    const float* w_gate = (const float*)d_in[1];   // [I,H]
    const float* w_up   = (const float*)d_in[2];   // [I,H]
    const float* w_down = (const float*)d_in[3];   // [H,I]
    float* out = (float*)d_out;                    // [B,S,H]

    // Workspace layout (bytes):
    //   xq  : M*H bf16  = 32 MiB   (A-fragment layout)
    //   wgq : I*H bf16  = 32 MiB   (B-fragment layout)
    //   wuq : I*H bf16  = 32 MiB
    //   wdq : H*I bf16  = 32 MiB
    //   h   : M*I bf16  = 128 MiB  (row-major)
    char* ws = (char*)d_ws;
    unsigned short* xq   = (unsigned short*)(ws);
    unsigned short* wgq  = (unsigned short*)(ws + (size_t)33554432);
    unsigned short* wuq  = (unsigned short*)(ws + (size_t)67108864);
    unsigned short* wdq  = (unsigned short*)(ws + (size_t)100663296);
    unsigned short* hbuf = (unsigned short*)(ws + (size_t)134217728);

    dim3 blk(256);

    // Quantize + pack all three weights (I*H/128 = 131072 groups each, 8 waves/block)
    int qblocks = (II * HH / GG) / 8;   // 16384
    quant_pack_w_kernel<<<qblocks, blk, 0, stream>>>(w_gate, wgq, II, HH);
    quant_pack_w_kernel<<<qblocks, blk, 0, stream>>>(w_up,   wuq, II, HH);
    quant_pack_w_kernel<<<qblocks, blk, 0, stream>>>(w_down, wdq, HH, II);

    // Pack x to bf16 A-fragments: M*H/8 threads
    pack_x_kernel<<<(MM * HH / 8) / 256, blk, 0, stream>>>(x, xq);

    // GEMM1: M x I over K=H, block tile 128x64
    gemm_gate_up_kernel<<<dim3(MM / 128, II / 64), blk, 0, stream>>>(xq, wgq, wuq, hbuf);

    // GEMM2: M x H over K=I, block tile 128x64
    gemm_down_kernel<<<dim3(MM / 128, HH / 64), blk, 0, stream>>>(hbuf, wdq, out);
}
